// TemporalGNN_78305843740934
// MI455X (gfx1250) — compile-verified, matrix operand order checked by reading, not run on hardware
//
#include <hip/hip_runtime.h>
#include <hip/hip_bf16.h>

// ---------------------------------------------------------------------------
// TemporalGNN for MI455X (gfx1250, wave32, WMMA)
//   T=12, N=50000, E=800000, F_IN=32, H=128, LSTM_H=64, N_CLS=8
// All GEMMs via v_wmma_f32_16x16x32_f16. h_seq is materialized in WMMA
// A-fragment order (layout shuffle staged through LDS, then coalesced
// global_store_b128) so the LSTM loads operands as contiguous v16h.
// Recurrent weights staged in LDS; LSTM state lives in VGPRs.
// ---------------------------------------------------------------------------

#define TT   12
#define NN   50000
#define EE   800000
#define FIN  32
#define HH   128
#define LH   64
#define NCLS 8
#define NTILES ((NN + 15) / 16)   // 3125 exactly

typedef __attribute__((ext_vector_type(16))) _Float16 v16h;
typedef __attribute__((ext_vector_type(8)))  float    v8f;
typedef _Float16 half_t;

union FragA {
  v16h v;
  unsigned int u[8];
  half_t h[16];
};

// K index inside a 32-wide K tile for 16-bit WMMA A/B operands (ISA 7.12.2):
// lane<16 : j=0..7 -> K=0..7,  j=8..15 -> K=16..23
// lane>=16: j=0..7 -> K=8..15, j=8..15 -> K=24..31
__device__ __forceinline__ int k_local_idx(int lane, int j) {
  return ((j < 8) ? j : (j + 8)) + ((lane >= 16) ? 8 : 0);
}

// Inverse map: K-within-tile kl -> (lane half, j slot)
__device__ __forceinline__ int frag_half(int kl) { return (kl >> 3) & 1; }
__device__ __forceinline__ int frag_j(int kl)    { return (kl & 7) + ((kl >= 16) ? 8 : 0); }

__device__ __forceinline__ float sigmoidf_fast(float x) {
  return 1.0f / (1.0f + __expf(-x));
}

// ---------------------------------------------------------------- degrees ---
__global__ void deg_count_kernel(const int* __restrict__ ei, float* __restrict__ deg) {
  int e = blockIdx.x * blockDim.x + threadIdx.x;
  if (e >= EE) return;
  atomicAdd(&deg[ei[EE + e]], 1.0f);
}

__global__ void dinv_kernel(const float* __restrict__ deg, float* __restrict__ dinv) {
  int n = blockIdx.x * blockDim.x + threadIdx.x;
  if (n >= NN) return;
  dinv[n] = rsqrtf(deg[n] + 1.0f);
}

// ----------------------------------------------------------- weight pack ----
// Packs B-fragments lane-contiguously: frag*512 + lane*16 + j (half_t).
__global__ void pack_kernel(const float* __restrict__ W_gcn, const float* __restrict__ b_gcn,
                            const float* __restrict__ gamma, const float* __restrict__ beta,
                            const float* __restrict__ mean, const float* __restrict__ var,
                            const float* __restrict__ W_ih, const float* __restrict__ W_hh,
                            const float* __restrict__ b_ih, const float* __restrict__ b_hh,
                            float* __restrict__ bgp, float* __restrict__ bias,
                            half_t* __restrict__ wg, half_t* __restrict__ wih,
                            half_t* __restrict__ whh) {
  int i = blockIdx.x * blockDim.x + threadIdx.x;
  if (i < HH) {  // folded GCN bias: (b - mean)*gamma*rsqrt(var+eps) + beta
    float sc = gamma[i] * rsqrtf(var[i] + 1e-5f);
    bgp[i] = (b_gcn[i] - mean[i]) * sc + beta[i];
    return;
  }
  i -= HH;
  if (i < 4 * LH) {  // LSTM bias, 256
    bias[i] = b_ih[i] + b_hh[i];
    return;
  }
  i -= 4 * LH;
  if (i < 8 * 512) {  // W_gcn' : 8 col-tiles of 32x16, BN scale folded
    int ct = i >> 9, rem = i & 511, lane = rem >> 4, j = rem & 15;
    int n = ct * 16 + (lane & 15);
    int k = k_local_idx(lane, j);
    float sc = gamma[n] * rsqrtf(var[n] + 1e-5f);
    wg[i] = (half_t)(W_gcn[k * HH + n] * sc);
    return;
  }
  i -= 8 * 512;
  if (i < 64 * 512) {  // W_ih^T : frag = kt*16 + ct (kt 0..3, ct 0..15)
    int frag = i >> 9, rem = i & 511, lane = rem >> 4, j = rem & 15;
    int kt = frag >> 4, ct = frag & 15;
    int r = ct * 16 + (lane & 15);
    int k = kt * 32 + k_local_idx(lane, j);
    wih[i] = (half_t)W_ih[r * HH + k];
    return;
  }
  i -= 64 * 512;
  if (i < 32 * 512) {  // W_hh^T : frag = kt*16 + ct (kt 0..1)
    int frag = i >> 9, rem = i & 511, lane = rem >> 4, j = rem & 15;
    int kt = frag >> 4, ct = frag & 15;
    int r = ct * 16 + (lane & 15);
    int k = kt * 32 + k_local_idx(lane, j);
    whh[i] = (half_t)W_hh[r * LH + k];
  }
}

// -------------------------------------------------------- agg init + scatter
__global__ void init_agg_kernel(const float* __restrict__ x, const float* __restrict__ dinv,
                                float* __restrict__ agg) {
  size_t i = (size_t)blockIdx.x * blockDim.x + threadIdx.x;  // over T*N rows
  if (i >= (size_t)TT * NN) return;
  int n = (int)(i % NN);
  float sn = dinv[n] * dinv[n];  // self-loop norm = 1/deg
  const float4* xr = (const float4*)(x + i * FIN);
  float4* ar = (float4*)(agg + i * FIN);
#pragma unroll
  for (int q = 0; q < FIN / 4; q++) {
    float4 v = xr[q];
    ar[q] = make_float4(sn * v.x, sn * v.y, sn * v.z, sn * v.w);
  }
}

__global__ void scatter_kernel(const float* __restrict__ x, const int* __restrict__ ei,
                               const float* __restrict__ dinv, float* __restrict__ agg) {
  int e = blockIdx.x * blockDim.x + threadIdx.x;
  if (e >= EE) return;
  int s = ei[e];
  int d = ei[EE + e];
  float w = dinv[s] * dinv[d];
#pragma unroll 1
  for (int t = 0; t < TT; t++) {
    const float4* xr = (const float4*)(x + ((size_t)t * NN + s) * FIN);
    float* ar = agg + ((size_t)t * NN + d) * FIN;
#pragma unroll
    for (int q = 0; q < FIN / 4; q++) {
      float4 v = xr[q];
      atomicAdd(ar + q * 4 + 0, w * v.x);
      atomicAdd(ar + q * 4 + 1, w * v.y);
      atomicAdd(ar + q * 4 + 2, w * v.z);
      atomicAdd(ar + q * 4 + 3, w * v.w);
    }
  }
}

// -------------------------------------------- GCN GEMM + BN(folded) + ReLU --
// One wave per 16-node tile: A = agg[16x32] f16, 8 WMMA -> [16x128].
// D-layout -> A-fragment inversion staged through LDS, then written to
// global as 4 contiguous v16h per lane (coalesced global_store_b128):
//   hfrag[((t*NTILES + tile)*4 + kt)*512 + frag_lane*16 + j]
__global__ void gcn_gemm_kernel(const float* __restrict__ agg, const half_t* __restrict__ wg,
                                const float* __restrict__ bgp, half_t* __restrict__ hfrag) {
  __shared__ half_t sOut[8][4 * 512];  // 32 KB: per-wave fragment staging
  const int wave = threadIdx.x >> 5, lane = threadIdx.x & 31;
  const int tile = blockIdx.x * 8 + wave;
  const int t = blockIdx.y;
  if (tile >= NTILES) return;  // uniform per wave: EXEC stays all-ones
  const int lrow = lane & 15, rowoff = (lane >= 16) ? 8 : 0;

  const float* arow = agg + ((size_t)t * NN + tile * 16 + lrow) * FIN;
  FragA a;
#pragma unroll
  for (int j = 0; j < 16; j += 2) {
    int k = k_local_idx(lane, j);  // j even -> k, k+1 consecutive
    a.h[j]     = (half_t)arow[k];
    a.h[j + 1] = (half_t)arow[k + 1];
  }

#pragma unroll
  for (int ct = 0; ct < 8; ct++) {
    v16h b = *(const v16h*)(wg + (size_t)ct * 512 + lane * 16);
    float bp = bgp[ct * 16 + lrow];
    v8f c;
#pragma unroll
    for (int r = 0; r < 8; r++) c[r] = bp;
    c = __builtin_amdgcn_wmma_f32_16x16x32_f16(false, a.v, false, b, (short)0, c,
                                               false, false);
    // LDS staging in A-fragment order: invert the K striping for this column
    int col = ct * 16 + lrow;
    int kt = col >> 5, kl = col & 31;
    int hf = frag_half(kl), j = frag_j(kl);
    half_t* fb = &sOut[wave][kt * 512 + j];
#pragma unroll
    for (int r = 0; r < 8; r++) {
      float vv = c[r];
      vv = vv > 0.0f ? vv : 0.0f;  // ReLU
      int frag_lane = (r + rowoff) + 16 * hf;  // row -> fragment lane
      fb[frag_lane * 16] = (half_t)vv;
    }
  }

  // coalesced writeout: per kt, lanes emit a linear 1 KB fragment burst
  // (same-wave DS ordering: no barrier needed between LDS store and load)
  half_t* tbase = hfrag + ((size_t)t * NTILES + tile) * 4 * 512;
#pragma unroll
  for (int kt = 0; kt < 4; kt++) {
    v16h v = *(const v16h*)&sOut[wave][kt * 512 + lane * 16];
    *(v16h*)(tbase + (size_t)kt * 512 + lane * 16) = v;
  }
}

// ------------------------------------------------- LSTM (12 steps) + head ---
// One wave per 16-node tile; 96 WMMA per step. x fragments load as contiguous
// v16h (b128 pairs); W_hh staged in LDS; h kept in fragment order in LDS so
// the recurrent repack is 2 contiguous v16h LDS reads.
__global__ void lstm_kernel(const half_t* __restrict__ hfrag, const half_t* __restrict__ wih,
                            const half_t* __restrict__ whh, const float* __restrict__ bias,
                            const float* __restrict__ W_cls, const float* __restrict__ b_cls,
                            float* __restrict__ out) {
  __shared__ half_t sWhh[32 * 512];     // 32 KB: all W_hh fragments
  __shared__ half_t sHf[8][2 * 512];    // 16 KB: per-wave h in fragment order
  const int wave = threadIdx.x >> 5, lane = threadIdx.x & 31;
  const int tile = blockIdx.x * 8 + wave;
  const int lrow = lane & 15, rowoff = (lane >= 16) ? 8 : 0;

  // cooperative W_hh -> LDS (before the tile guard: barrier is uniform)
  for (int i = threadIdx.x; i < (32 * 512) / 8; i += 256)
    ((float4*)sWhh)[i] = ((const float4*)whh)[i];
  __syncthreads();
  if (tile >= NTILES) return;

  // hoist per-lane biases: bb[g4*4+ut]
  float bb[16];
#pragma unroll
  for (int g4 = 0; g4 < 4; g4++)
#pragma unroll
    for (int ut = 0; ut < 4; ut++) bb[g4 * 4 + ut] = bias[g4 * 64 + ut * 16 + lrow];

  FragA hA[2];
  v8f cst[4];
#pragma unroll
  for (int q = 0; q < 2; q++)
#pragma unroll
    for (int u = 0; u < 8; u++) hA[q].u[u] = 0u;
#pragma unroll
  for (int u = 0; u < 4; u++)
#pragma unroll
    for (int r = 0; r < 8; r++) cst[u][r] = 0.0f;

  for (int t = 0; t < TT; t++) {
    // x = h_seq[t] tile, already in A-fragment order: 4 contiguous v16h loads
    const half_t* tbase = hfrag + ((size_t)t * NTILES + tile) * 4 * 512;
    FragA xA[4];
#pragma unroll
    for (int kt = 0; kt < 4; kt++)
      xA[kt].v = *(const v16h*)(tbase + (size_t)kt * 512 + lane * 16);

#pragma unroll
    for (int ut = 0; ut < 4; ut++) {  // hidden-unit col tile (16 units)
      v8f acc[4];                     // i, f, g, o accumulators
#pragma unroll
      for (int g4 = 0; g4 < 4; g4++) {
        int ct = g4 * 4 + ut;  // col tile inside the 256-wide gate matrix
        float b0 = bb[g4 * 4 + ut];
#pragma unroll
        for (int r = 0; r < 8; r++) acc[g4][r] = b0;
#pragma unroll
        for (int kt = 0; kt < 4; kt++) {  // input contribution, K=128
          v16h b = *(const v16h*)(wih + (size_t)(kt * 16 + ct) * 512 + lane * 16);
          acc[g4] = __builtin_amdgcn_wmma_f32_16x16x32_f16(false, xA[kt].v, false, b,
                                                           (short)0, acc[g4], false, false);
        }
#pragma unroll
        for (int kt = 0; kt < 2; kt++) {  // recurrent contribution, K=64 (LDS)
          v16h b = *(const v16h*)(sWhh + (kt * 16 + ct) * 512 + lane * 16);
          acc[g4] = __builtin_amdgcn_wmma_f32_16x16x32_f16(false, hA[kt].v, false, b,
                                                           (short)0, acc[g4], false, false);
        }
      }
      // elementwise LSTM cell update; write h in fragment order into LDS
      int col = ut * 16 + lrow;
      int kt = col >> 5, kl = col & 31;
      int hf = frag_half(kl), j = frag_j(kl);
      half_t* fb = &sHf[wave][kt * 512 + j];
#pragma unroll
      for (int r = 0; r < 8; r++) {
        float ig = sigmoidf_fast(acc[0][r]);
        float fg = sigmoidf_fast(acc[1][r]);
        float gg = tanhf(acc[2][r]);
        float og = sigmoidf_fast(acc[3][r]);
        float cv = fg * cst[ut][r] + ig * gg;
        cst[ut][r] = cv;
        float hv = og * tanhf(cv);
        int frag_lane = (r + rowoff) + 16 * hf;
        fb[frag_lane * 16] = (half_t)hv;
      }
    }
    // recurrent repack: h fragments are contiguous in LDS now (per-wave,
    // same-wave DS ops are in order -> no barrier needed)
#pragma unroll
    for (int kt = 0; kt < 2; kt++)
      hA[kt].v = *(const v16h*)&sHf[wave][kt * 512 + lane * 16];
  }

  // classifier + log_softmax: lanes 0..15 each own one node row
  if (lane < 16) {
    float logit[NCLS];
#pragma unroll
    for (int cc = 0; cc < NCLS; cc++) logit[cc] = b_cls[cc];
    for (int k = 0; k < LH; k++) {  // read h[lane][k] from fragment order
      int kt = k >> 5, kl = k & 31;
      int fl = lane + 16 * frag_half(kl);
      float hv = (float)sHf[wave][kt * 512 + fl * 16 + frag_j(kl)];
#pragma unroll
      for (int cc = 0; cc < NCLS; cc++) logit[cc] += hv * W_cls[k * NCLS + cc];
    }
    float m = logit[0];
#pragma unroll
    for (int cc = 1; cc < NCLS; cc++) m = fmaxf(m, logit[cc]);
    float s = 0.0f;
#pragma unroll
    for (int cc = 0; cc < NCLS; cc++) s += __expf(logit[cc] - m);
    float ls = m + __logf(s);
    int node = tile * 16 + lane;
#pragma unroll
    for (int cc = 0; cc < NCLS; cc++) out[(size_t)node * NCLS + cc] = logit[cc] - ls;
  }
}

// ---------------------------------------------------------------- launch ----
extern "C" void kernel_launch(void* const* d_in, const int* in_sizes, int n_in,
                              void* d_out, int out_size, void* d_ws, size_t ws_size,
                              hipStream_t stream) {
  const float* x      = (const float*)d_in[0];
  const int*   ei     = (const int*)d_in[1];
  const float* W_gcn  = (const float*)d_in[2];
  const float* b_gcn  = (const float*)d_in[3];
  const float* gamma  = (const float*)d_in[4];
  const float* beta   = (const float*)d_in[5];
  const float* mean   = (const float*)d_in[6];
  const float* var    = (const float*)d_in[7];
  const float* W_ih   = (const float*)d_in[8];
  const float* W_hh   = (const float*)d_in[9];
  const float* b_ih   = (const float*)d_in[10];
  const float* b_hh   = (const float*)d_in[11];
  const float* W_cls  = (const float*)d_in[12];
  const float* b_cls  = (const float*)d_in[13];
  float* out = (float*)d_out;

  char* ws = (char*)d_ws;
  size_t off = 0;
  auto take = [&](size_t bytes) {
    size_t o = off;
    off += (bytes + 255) & ~(size_t)255;
    return o;
  };
  float*  deg   = (float*)(ws + take((size_t)NN * 4));
  float*  dinv  = (float*)(ws + take((size_t)NN * 4));
  float*  bgp   = (float*)(ws + take((size_t)HH * 4));
  float*  bias  = (float*)(ws + take((size_t)4 * LH * 4));
  half_t* wg    = (half_t*)(ws + take((size_t)8 * 512 * 2));
  half_t* wih   = (half_t*)(ws + take((size_t)64 * 512 * 2));
  half_t* whh   = (half_t*)(ws + take((size_t)32 * 512 * 2));
  float*  agg   = (float*)(ws + take((size_t)TT * NN * FIN * 4));
  half_t* hfrag = (half_t*)(ws + take((size_t)TT * NN * HH * 2));
  (void)ws_size; (void)in_sizes; (void)n_in; (void)out_size;

  hipMemsetAsync(deg, 0, (size_t)NN * 4, stream);
  deg_count_kernel<<<(EE + 255) / 256, 256, 0, stream>>>(ei, deg);
  dinv_kernel<<<(NN + 255) / 256, 256, 0, stream>>>(deg, dinv);

  const int pack_items = HH + 4 * LH + 8 * 512 + 64 * 512 + 32 * 512;
  pack_kernel<<<(pack_items + 255) / 256, 256, 0, stream>>>(
      W_gcn, b_gcn, gamma, beta, mean, var, W_ih, W_hh, b_ih, b_hh,
      bgp, bias, wg, wih, whh);

  const size_t rows = (size_t)TT * NN;
  init_agg_kernel<<<(unsigned)((rows + 255) / 256), 256, 0, stream>>>(x, dinv, agg);
  scatter_kernel<<<(EE + 255) / 256, 256, 0, stream>>>(x, ei, dinv, agg);

  dim3 ggrid((NTILES + 7) / 8, TT);
  gcn_gemm_kernel<<<ggrid, 256, 0, stream>>>(agg, wg, bgp, hfrag);

  lstm_kernel<<<(NTILES + 7) / 8, 256, 0, stream>>>(hfrag, wih, whh, bias, W_cls, b_cls, out);
}
